// TiConv2d_39548058861899
// MI455X (gfx1250) — 8-bit matrix operand order repaired and checked by reading, compiled, not run on hardware
//
#include <hip/hip_runtime.h>
#include <hip/hip_bf16.h>

typedef __attribute__((ext_vector_type(8))) int v8i;

#define TI_N    32
#define TI_H    64
#define TI_W    64
#define TI_C    128
#define TI_TOT  (TI_N * TI_H * TI_W * TI_C)
#define TI_WGT  (TI_C * 3 * 3 * TI_C)
#define ACC_BYTES ((size_t)TI_TOT * 4)
#define ACT_BYTES ((size_t)TI_TOT)
#define WGT_BYTES ((size_t)TI_WGT)

__global__ void ti_init_kernel(unsigned* gmax) { *gmax = 0u; }

__global__ void ti_pack_kernel(const int* __restrict__ in, unsigned* __restrict__ out, int n4) {
    int i = blockIdx.x * blockDim.x + threadIdx.x;
    if (i < n4) {
        int4 v = ((const int4*)in)[i];
        unsigned p = (v.x & 0xff) | ((v.y & 0xff) << 8) |
                     ((v.z & 0xff) << 16) | ((unsigned)(v.w & 0xff) << 24);
        out[i] = p;
    }
}

__global__ void __launch_bounds__(256)
ti_conv_wmma_kernel(const unsigned char* __restrict__ actp,
                    const unsigned char* __restrict__ wp,
                    int* __restrict__ acc,
                    unsigned* __restrict__ gmax) {
    __shared__ int4 lds_act4[432];
    unsigned char* lds_act = (unsigned char*)lds_act4;

    const int tid  = threadIdx.x;
    const int lane = tid & 31;
    const int half = lane >> 4;
    const int mrow = lane & 15;

    const int pix_base = blockIdx.x << 4;
    const int wb = pix_base & (TI_W - 1);
    const int hh = (pix_base >> 6) & (TI_H - 1);
    const int nn = pix_base >> 12;
    const int co_base = (tid >> 5) << 4;

    const unsigned char* act_n = actp + ((size_t)nn << 19);
#pragma unroll
    for (int it = 0; it < 2; ++it) {
        const int q = tid + it * 256;
        if (q < 432) {
            const int off16 = (q & 7) << 4;
            const int seg = q >> 3;
            const int col = seg % 18;
            const int row = seg / 18;
            const int h_in = hh + row - 1;
            const int w_in = wb + col - 1;
            const unsigned lds_off = (unsigned)(uintptr_t)(&lds_act[q << 4]);
            if (h_in >= 0 && h_in < TI_H && w_in >= 0 && w_in < TI_W) {
                const unsigned goff =
                    (unsigned)((((h_in << 6) + w_in) << 7) + off16);
                asm volatile("global_load_async_to_lds_b128 %0, %1, %2"
                             :: "v"(lds_off), "v"(goff), "s"(act_n)
                             : "memory");
            } else {
                int4 z = {0, 0, 0, 0};
                *(int4*)(&lds_act[q << 4]) = z;
            }
        }
    }
    asm volatile("s_wait_asynccnt 0x0" ::: "memory");
    __syncthreads();

    v8i c = {0, 0, 0, 0, 0, 0, 0, 0};

#pragma unroll
    for (int kh = 0; kh < 3; ++kh) {
#pragma unroll
        for (int kw = 0; kw < 3; ++kw) {
            const int abase = ((kh * 18) + (mrow + kw)) << 7;
            const unsigned char* wrow =
                wp + (((size_t)((co_base + mrow) * 9 + kh * 3 + kw)) << 7);
#pragma unroll
            for (int kk = 0; kk < 2; ++kk) {
                const int ab = abase + (kk << 6) + (half << 4);
                v8i a;
#pragma unroll
                for (int j = 0; j < 4; ++j) {
                    int2 d = *(const int2*)(&lds_act[ab + ((j >> 1) << 5) + ((j & 1) << 3)]);
                    a[2 * j]     = d.x;
                    a[2 * j + 1] = d.y;
                }
                const unsigned char* bp = wrow + (kk << 6) + (half << 4);
                int4 b0 = *(const int4*)(bp);
                int4 b1 = *(const int4*)(bp + 32);
                v8i b;
                b[0] = b0.x; b[1] = b0.y; b[2] = b0.z; b[3] = b0.w;
                b[4] = b1.x; b[5] = b1.y; b[6] = b1.z; b[7] = b1.w;
                c = __builtin_amdgcn_wmma_i32_16x16x64_iu8(true, a, true, b, c,
                                                           false, false);
            }
        }
    }

    unsigned mx = 0u;
#pragma unroll
    for (int r = 0; r < 8; ++r) {
        const int v = c[r];
        const unsigned av = (unsigned)(v < 0 ? -v : v);
        mx = mx > av ? mx : av;
        const int pix = pix_base + (half << 3) + r;
        acc[(size_t)pix * TI_C + co_base + mrow] = v;
    }
#pragma unroll
    for (int off = 16; off > 0; off >>= 1) {
        unsigned o = (unsigned)__shfl_xor((int)mx, off, 32);
        mx = mx > o ? mx : o;
    }
    if (lane == 0) atomicMax(gmax, mx);
}

__device__ __forceinline__ float ti_requant1(int x, int shift, int s) {
    const int q = x >> s;
    const int rem = x - (q << s);
    const int rd = rem >> (s - 1);
    int r = q + rd;
    r = r > 127 ? 127 : (r < -127 ? -127 : r);
    const signed char res = (shift > 0) ? (signed char)r : (signed char)x;
    return (float)res;
}

__global__ void ti_requant_kernel(const int* __restrict__ acc,
                                  const unsigned* __restrict__ gmax,
                                  const int* __restrict__ e_act,
                                  const int* __restrict__ e_w,
                                  float* __restrict__ out, int total) {
    const int i = blockIdx.x * blockDim.x + threadIdx.x;
    const unsigned m = *gmax;
    const int bw = (m <= 1u) ? 0 : (32 - __clz((int)(m - 1u)));
    const int shift = bw - 7;
    const int s = shift > 1 ? shift : 1;
    const int base = i << 2;
    if (base < total) {
        int4 x4 = *(const int4*)(acc + base);
        float4 o;
        o.x = ti_requant1(x4.x, shift, s);
        o.y = ti_requant1(x4.y, shift, s);
        o.z = ti_requant1(x4.z, shift, s);
        o.w = ti_requant1(x4.w, shift, s);
        *(float4*)(out + base) = o;
    }
    if (i == 0) {
        out[total] = (float)(*e_act + *e_w + (shift > 0 ? shift : 0));
    }
}

extern "C" void kernel_launch(void* const* d_in, const int* in_sizes, int n_in,
                              void* d_out, int out_size, void* d_ws, size_t ws_size,
                              hipStream_t stream) {
    const int* act   = (const int*)d_in[0];
    const int* wgt   = (const int*)d_in[1];
    const int* e_act = (const int*)d_in[2];
    const int* e_w   = (const int*)d_in[3];
    float* out = (float*)d_out;

    char* ws = (char*)d_ws;
    int* acc = (int*)ws;
    unsigned char* actp = (unsigned char*)(ws + ACC_BYTES);
    unsigned char* wp   = actp + ACT_BYTES;
    unsigned* gmax = (unsigned*)(wp + WGT_BYTES);

    ti_init_kernel<<<1, 1, 0, stream>>>(gmax);

    const int act4 = TI_TOT / 4;
    ti_pack_kernel<<<(act4 + 255) / 256, 256, 0, stream>>>(act, (unsigned*)actp, act4);
    const int wgt4 = TI_WGT / 4;
    ti_pack_kernel<<<(wgt4 + 255) / 256, 256, 0, stream>>>(wgt, (unsigned*)wp, wgt4);

    ti_conv_wmma_kernel<<<8192, 256, 0, stream>>>(actp, wp, acc, gmax);

    const int rq_threads = TI_TOT / 4;
    ti_requant_kernel<<<(rq_threads + 255) / 256, 256, 0, stream>>>(acc, gmax, e_act,
                                                                    e_w, out, TI_TOT);
}
